// VectorQuantizer_1726576853954
// MI455X (gfx1250) — matrix operand order repaired and checked by reading, not compiled, not run on hardware
//
#include <hip/hip_runtime.h>
#include <hip/hip_bf16.h>

#define N_E   8192
#define E_DIM 512
#define NROWS 8192   // B*H*W = 32*16*16

typedef __attribute__((ext_vector_type(16))) __bf16 v16bf;
typedef __attribute__((ext_vector_type(8)))  __bf16 v8bf;
typedef __attribute__((ext_vector_type(8)))  float  v8f;
typedef __attribute__((ext_vector_type(4)))  float  v4f;

// ---------------------------------------------------------------------------
// K1: row norms + bf16 hi/lo split of z and normalized codebook; zero accums.
// Blocks 0..8191: codebook rows.  Blocks 8192..16383: z rows.
// ---------------------------------------------------------------------------
__global__ __launch_bounds__(256) void vq_prep(
    const float* __restrict__ z, const float* __restrict__ emb,
    __bf16* __restrict__ zhi, __bf16* __restrict__ zlo,
    __bf16* __restrict__ cbhi, __bf16* __restrict__ cblo,
    float* __restrict__ zn, float* __restrict__ cbn, float* __restrict__ invn,
    unsigned* __restrict__ hist, float* __restrict__ cos_sum)
{
  __shared__ float red[256];
  const int b = blockIdx.x;
  const int t = threadIdx.x;
  const bool isEmb = (b < N_E);
  const int row = isEmb ? b : (b - N_E);
  const float* src = (isEmb ? emb : z) + (size_t)row * E_DIM;
  const float x0 = src[t];
  const float x1 = src[t + 256];
  red[t] = x0 * x0 + x1 * x1;
  __syncthreads();
  for (int s = 128; s > 0; s >>= 1) {
    if (t < s) red[t] += red[t + s];
    __syncthreads();
  }
  const float sumsq = red[0];
  __syncthreads();

  const size_t o = (size_t)row * E_DIM + t;
  if (isEmb) {
    const float inv = 1.0f / sqrtf(sumsq);          // cb = emb / ||emb||
    const float c0 = x0 * inv, c1 = x1 * inv;
    red[t] = c0 * c0 + c1 * c1;                     // cb_norm (≈1, match ref)
    __syncthreads();
    for (int s = 128; s > 0; s >>= 1) {
      if (t < s) red[t] += red[t + s];
      __syncthreads();
    }
    if (t == 0) { invn[row] = inv; cbn[row] = sqrtf(red[0]); }
    const __bf16 h0 = (__bf16)c0;
    const __bf16 h1 = (__bf16)c1;
    cbhi[o]       = h0; cblo[o]       = (__bf16)(c0 - (float)h0);
    cbhi[o + 256] = h1; cblo[o + 256] = (__bf16)(c1 - (float)h1);
  } else {
    if (t == 0) zn[row] = sqrtf(sumsq);
    const __bf16 h0 = (__bf16)x0;
    const __bf16 h1 = (__bf16)x1;
    zhi[o]       = h0; zlo[o]       = (__bf16)(x0 - (float)h0);
    zhi[o + 256] = h1; zlo[o + 256] = (__bf16)(x1 - (float)h1);
  }
  if (b < 32) hist[b * 256 + t] = 0u;
  if (b == 0 && t == 0) *cos_sum = 0.0f;
}

// ---------------------------------------------------------------------------
// K2: d = (z @ cb^T) / (||z||*||cb|| + 1e-6) via bf16x3 WMMA emulation.
// Block = 64 rows x 128 cols of d; 8 waves (2 M x 4 N); each wave = 32x32
// (2x2 f32 accumulators; 12 WMMAs per K-step from 8 fragment loads).
//
// Fragment layouts used here (wave32; A and B share contiguous k order):
//  A 16x32 bf16 : lane L holds row M=L%16; element e -> K = e + (L>=16?16:0)
//                 (two 16B loads per lane)
//  B 32x16 bf16 : lane L holds col N=L%16; element e -> K = e + (L>=16?16:0)
//                 (one 32B contiguous load per lane, cb stored row-major)
//  C/D 16x16 f32: lane L col N=L%16; VGPR r -> row M = r + (L>=16?8:0)
// ---------------------------------------------------------------------------
__global__ __launch_bounds__(256) void vq_gemm(
    const __bf16* __restrict__ zhi, const __bf16* __restrict__ zlo,
    const __bf16* __restrict__ cbhi, const __bf16* __restrict__ cblo,
    const float* __restrict__ zn, const float* __restrict__ cbn,
    float* __restrict__ d_mat)
{
  const int tid  = threadIdx.x;
  const int wave = tid >> 5;
  const int lane = tid & 31;
  const int half = lane >> 4;
  const int l16  = lane & 15;
  const int rowBase = blockIdx.y * 64  + (wave & 1) * 32;   // 2 M subtiles
  const int colBase = blockIdx.x * 128 + (wave >> 1) * 32;  // 4 N subtiles

  const size_t arow0 = (size_t)(rowBase + l16) * E_DIM;
  const size_t arow1 = (size_t)(rowBase + 16 + l16) * E_DIM;
  const int    aoff  = half * 16;
  const int    bj0   = colBase + l16;
  const int    bj1   = bj0 + 16;
  const size_t brow0 = (size_t)bj0 * E_DIM;
  const size_t brow1 = (size_t)bj1 * E_DIM;
  const int    boff  = half * 16;

  v8f acc00 = {}, acc01 = {}, acc10 = {}, acc11 = {};
  for (int k0 = 0; k0 < E_DIM; k0 += 32) {
    // A fragments (hi/lo), rows rowBase..+15 and rowBase+16..+31
    const v8bf a0h0 = *(const v8bf*)(zhi + arow0 + k0 + aoff);
    const v8bf a0h1 = *(const v8bf*)(zhi + arow0 + k0 + 8 + aoff);
    const v8bf a0l0 = *(const v8bf*)(zlo + arow0 + k0 + aoff);
    const v8bf a0l1 = *(const v8bf*)(zlo + arow0 + k0 + 8 + aoff);
    const v8bf a1h0 = *(const v8bf*)(zhi + arow1 + k0 + aoff);
    const v8bf a1h1 = *(const v8bf*)(zhi + arow1 + k0 + 8 + aoff);
    const v8bf a1l0 = *(const v8bf*)(zlo + arow1 + k0 + aoff);
    const v8bf a1l1 = *(const v8bf*)(zlo + arow1 + k0 + 8 + aoff);
    const v16bf a0h = __builtin_shufflevector(a0h0, a0h1, 0,1,2,3,4,5,6,7,8,9,10,11,12,13,14,15);
    const v16bf a0l = __builtin_shufflevector(a0l0, a0l1, 0,1,2,3,4,5,6,7,8,9,10,11,12,13,14,15);
    const v16bf a1h = __builtin_shufflevector(a1h0, a1h1, 0,1,2,3,4,5,6,7,8,9,10,11,12,13,14,15);
    const v16bf a1l = __builtin_shufflevector(a1l0, a1l1, 0,1,2,3,4,5,6,7,8,9,10,11,12,13,14,15);
    // B fragments (hi/lo), cols colBase..+15 and colBase+16..+31
    const v16bf b0h = *(const v16bf*)(cbhi + brow0 + k0 + boff);
    const v16bf b0l = *(const v16bf*)(cblo + brow0 + k0 + boff);
    const v16bf b1h = *(const v16bf*)(cbhi + brow1 + k0 + boff);
    const v16bf b1l = *(const v16bf*)(cblo + brow1 + k0 + boff);
    // d ≈ hi*hi + hi*lo + lo*hi  (lo*lo ~ 2^-32, dropped)
    acc00 = __builtin_amdgcn_wmma_f32_16x16x32_bf16(false, a0h, false, b0h, (short)0, acc00, false, false);
    acc01 = __builtin_amdgcn_wmma_f32_16x16x32_bf16(false, a0h, false, b1h, (short)0, acc01, false, false);
    acc10 = __builtin_amdgcn_wmma_f32_16x16x32_bf16(false, a1h, false, b0h, (short)0, acc10, false, false);
    acc11 = __builtin_amdgcn_wmma_f32_16x16x32_bf16(false, a1h, false, b1h, (short)0, acc11, false, false);
    acc00 = __builtin_amdgcn_wmma_f32_16x16x32_bf16(false, a0h, false, b0l, (short)0, acc00, false, false);
    acc01 = __builtin_amdgcn_wmma_f32_16x16x32_bf16(false, a0h, false, b1l, (short)0, acc01, false, false);
    acc10 = __builtin_amdgcn_wmma_f32_16x16x32_bf16(false, a1h, false, b0l, (short)0, acc10, false, false);
    acc11 = __builtin_amdgcn_wmma_f32_16x16x32_bf16(false, a1h, false, b1l, (short)0, acc11, false, false);
    acc00 = __builtin_amdgcn_wmma_f32_16x16x32_bf16(false, a0l, false, b0h, (short)0, acc00, false, false);
    acc01 = __builtin_amdgcn_wmma_f32_16x16x32_bf16(false, a0l, false, b1h, (short)0, acc01, false, false);
    acc10 = __builtin_amdgcn_wmma_f32_16x16x32_bf16(false, a1l, false, b0h, (short)0, acc10, false, false);
    acc11 = __builtin_amdgcn_wmma_f32_16x16x32_bf16(false, a1l, false, b1h, (short)0, acc11, false, false);
  }
  const float cb0 = cbn[bj0];
  const float cb1 = cbn[bj1];
#pragma unroll
  for (int r = 0; r < 8; ++r) {
    const int m0 = rowBase + r + half * 8;
    const int m1 = m0 + 16;
    const float zn0 = zn[m0];
    const float zn1 = zn[m1];
    const size_t o0 = (size_t)m0 * N_E;
    const size_t o1 = (size_t)m1 * N_E;
    d_mat[o0 + bj0] = acc00[r] / (zn0 * cb0 + 1e-6f);
    d_mat[o0 + bj1] = acc01[r] / (zn0 * cb1 + 1e-6f);
    d_mat[o1 + bj0] = acc10[r] / (zn1 * cb0 + 1e-6f);
    d_mat[o1 + bj1] = acc11[r] / (zn1 * cb1 + 1e-6f);
  }
}

// ---------------------------------------------------------------------------
// K3: per row of d (cached in LDS): argmax (first-occurrence), softmax,
// one-hot, histogram, cosine-loss contribution.  prob/onehot are write-once
// streams -> non-temporal stores so they don't evict d from L2.
// ---------------------------------------------------------------------------
__global__ __launch_bounds__(256) void vq_softmax(
    const float* __restrict__ d_mat, const float* __restrict__ zn,
    const float* __restrict__ cbn,
    float* __restrict__ prob, float* __restrict__ onehot, float* __restrict__ idx_f,
    int* __restrict__ ridx_out, unsigned* __restrict__ hist, float* __restrict__ cos_sum)
{
  __shared__ float row[N_E];          // 32 KB
  __shared__ float sval[256];
  __shared__ int   sidx[256];
  const int i = blockIdx.x;
  const int t = threadIdx.x;
  const v4f* src = (const v4f*)(d_mat + (size_t)i * N_E);
  v4f* rowv = (v4f*)row;

  float best = -3.402823e38f;
  int bidx = 0;
#pragma unroll
  for (int c = 0; c < 8; ++c) {
    const int j4 = t + 256 * c;
    const v4f v = src[j4];
    rowv[j4] = v;
    const int j = j4 * 4;
    if (v.x > best) { best = v.x; bidx = j; }
    if (v.y > best) { best = v.y; bidx = j + 1; }
    if (v.z > best) { best = v.z; bidx = j + 2; }
    if (v.w > best) { best = v.w; bidx = j + 3; }
  }
  sval[t] = best; sidx[t] = bidx;
  __syncthreads();
  for (int s = 128; s > 0; s >>= 1) {
    if (t < s) {
      const float ov = sval[t + s]; const int oi = sidx[t + s];
      if (ov > sval[t] || (ov == sval[t] && oi < sidx[t])) { sval[t] = ov; sidx[t] = oi; }
    }
    __syncthreads();
  }
  const float rmax = sval[0];
  const int   rj   = sidx[0];
  __syncthreads();

  float acc = 0.0f;
#pragma unroll
  for (int c = 0; c < 8; ++c) {
    const v4f v = rowv[t + 256 * c];
    acc += expf(v.x - rmax) + expf(v.y - rmax) + expf(v.z - rmax) + expf(v.w - rmax);
  }
  sval[t] = acc;
  __syncthreads();
  for (int s = 128; s > 0; s >>= 1) {
    if (t < s) sval[t] += sval[t + s];
    __syncthreads();
  }
  const float inv = 1.0f / sval[0];

  v4f* pp = (v4f*)(prob   + (size_t)i * N_E);
  v4f* po = (v4f*)(onehot + (size_t)i * N_E);
#pragma unroll
  for (int c = 0; c < 8; ++c) {
    const int j4 = t + 256 * c;
    const v4f v = rowv[j4];
    v4f p;
    p.x = expf(v.x - rmax) * inv;
    p.y = expf(v.y - rmax) * inv;
    p.z = expf(v.z - rmax) * inv;
    p.w = expf(v.w - rmax) * inv;
    __builtin_nontemporal_store(p, pp + j4);
    const int j = j4 * 4;
    v4f oh;
    oh.x = (j     == rj) ? 1.0f : 0.0f;
    oh.y = (j + 1 == rj) ? 1.0f : 0.0f;
    oh.z = (j + 2 == rj) ? 1.0f : 0.0f;
    oh.w = (j + 3 == rj) ? 1.0f : 0.0f;
    __builtin_nontemporal_store(oh, po + j4);
  }
  if (t == 0) {
    ridx_out[i] = rj;
    __builtin_nontemporal_store((float)rj, idx_f + i);
    atomicAdd(&hist[rj], 1u);
    // cos(z_q_i, z_i): dot = d[i,rj] * (zn*cbn + 1e-6); norms (z_q)=cbn, (z)=zn
    const float zni = zn[i];
    const float cbj = cbn[rj];
    const float dot = rmax * (zni * cbj + 1e-6f);
    const float cv  = dot / (fmaxf(cbj, 1e-8f) * fmaxf(zni, 1e-8f));
    atomicAdd(cos_sum, cv);
  }
}

// ---------------------------------------------------------------------------
// K4: z_q_st[i,:] = z[i,:] + (cb[idx[i],:] - z[i,:])   (write-once -> NT)
// ---------------------------------------------------------------------------
__global__ __launch_bounds__(128) void vq_zq(
    const float* __restrict__ z, const float* __restrict__ emb,
    const float* __restrict__ invn, const int* __restrict__ ridx,
    float* __restrict__ zq_st)
{
  const int i = blockIdx.x;
  const int t = threadIdx.x;
  const int j = ridx[i];
  const float inv = invn[j];
  const v4f e  = ((const v4f*)(emb + (size_t)j * E_DIM))[t];
  const v4f zv = ((const v4f*)(z   + (size_t)i * E_DIM))[t];
  v4f r;
  r.x = zv.x + (e.x * inv - zv.x);
  r.y = zv.y + (e.y * inv - zv.y);
  r.z = zv.z + (e.z * inv - zv.z);
  r.w = zv.w + (e.w * inv - zv.w);
  __builtin_nontemporal_store(r, (v4f*)(zq_st + (size_t)i * E_DIM) + t);
}

// ---------------------------------------------------------------------------
// K5: scalars: loss, loss_kl, perplexity.
// loss_kl = log(N_E) + sum(e*log(e+1e-6));  perplexity = exp(-sum(e*log(e+1e-6)))
// loss    = 1.25 * (1 - mean(cos))   (both cosine terms equal in fwd value)
// ---------------------------------------------------------------------------
__global__ __launch_bounds__(256) void vq_finalize(
    const unsigned* __restrict__ hist, const float* __restrict__ cos_sum,
    float* __restrict__ out, float* __restrict__ o_perp)
{
  __shared__ float red[256];
  const int t = threadIdx.x;
  float s = 0.0f;
  for (int c = 0; c < 32; ++c) {
    const float e = (float)hist[t + 256 * c] * (1.0f / (float)NROWS);
    s += e * logf(e + 1e-6f);
  }
  red[t] = s;
  __syncthreads();
  for (int ss = 128; ss > 0; ss >>= 1) {
    if (t < ss) red[t] += red[t + ss];
    __syncthreads();
  }
  if (t == 0) {
    const float s1 = red[0];
    const float mc = (*cos_sum) * (1.0f / (float)NROWS);
    out[0] = (1.0f - mc) + 0.25f * (1.0f - mc);   // loss
    out[1] = 0.0f;                                // constrative_loss
    out[2] = logf((float)N_E) + s1;               // loss_kl
    *o_perp = expf(-s1);                          // perplexity
  }
}

// ---------------------------------------------------------------------------
extern "C" void kernel_launch(void* const* d_in, const int* in_sizes, int n_in,
                              void* d_out, int out_size, void* d_ws, size_t ws_size,
                              hipStream_t stream)
{
  const float* z   = (const float*)d_in[0];   // (32,16,16,512) -> (8192,512)
  const float* emb = (const float*)d_in[1];   // (8192,512)
  float* out = (float*)d_out;

  // Output layout (flat, return order):
  // [0] loss, [1] constrative, [2] loss_kl, prob(64Mi), d(64Mi),
  // z_q_st(4Mi), perplexity(1), min_encodings(64Mi), indices(8192)
  const size_t SZ_DD = (size_t)NROWS * N_E;
  float* o_prob   = out + 3;
  float* o_d      = o_prob + SZ_DD;
  float* o_zq     = o_d + SZ_DD;
  float* o_perp   = o_zq + (size_t)NROWS * E_DIM;
  float* o_onehot = o_perp + 1;
  float* o_idx    = o_onehot + SZ_DD;

  // Workspace layout (~33.7 MB)
  const size_t BF = (size_t)NROWS * E_DIM;    // 4 Mi elements
  __bf16* zhi  = (__bf16*)d_ws;
  __bf16* zlo  = zhi + BF;
  __bf16* cbhi = zlo + BF;
  __bf16* cblo = cbhi + BF;
  float*  zn   = (float*)(cblo + BF);
  float*  cbn  = zn + NROWS;
  float*  invn = cbn + N_E;
  int*    ridx = (int*)(invn + N_E);
  unsigned* hist = (unsigned*)(ridx + NROWS);
  float*  cos_sum = (float*)(hist + N_E);

  vq_prep<<<dim3(2 * N_E), dim3(256), 0, stream>>>(
      z, emb, zhi, zlo, cbhi, cblo, zn, cbn, invn, hist, cos_sum);
  vq_gemm<<<dim3(N_E / 128, NROWS / 64), dim3(256), 0, stream>>>(
      zhi, zlo, cbhi, cblo, zn, cbn, o_d);
  vq_softmax<<<dim3(NROWS), dim3(256), 0, stream>>>(
      o_d, zn, cbn, o_prob, o_onehot, o_idx, ridx, hist, cos_sum);
  vq_zq<<<dim3(NROWS), dim3(128), 0, stream>>>(z, emb, invn, ridx, o_zq);
  vq_finalize<<<dim3(1), dim3(256), 0, stream>>>(hist, cos_sum, out, o_perp);
}
